// KANCFeatureExtractor_23295902613663
// MI455X (gfx1250) — matrix operand order repaired and checked by reading, not compiled, not run on hardware
//
#include <hip/hip_runtime.h>
#include <hip/hip_bf16.h>

// ---------------------------------------------------------------------------
// KAN-Conv feature extractor for MI455X (gfx1250, wave32).
//   conv layers: VALU (tiny N=5/2, heavy nonlinear per-element work)
//   FC layer:    v_wmma_f32_16x16x32_f16 on pre-packed, zero-padded f16 tiles
// ---------------------------------------------------------------------------

typedef __attribute__((ext_vector_type(16))) _Float16 v16h;
typedef __attribute__((ext_vector_type(8)))  _Float16 v8h;
typedef __attribute__((ext_vector_type(8)))  float    v8f;

// Uniform cubic B-spline over grid [-2.2, 2.2], h = 0.4 (G=5, K=3).
// For x in interval i (i = floor((x+2.2)/0.4), valid 0..10), the 4 nonzero
// order-3 bases sit at output indices j = i-3 .. i (kept only if 0<=j<8),
// with the standard uniform-cubic weights in local coordinate u.
template<int CIN, int KS, int HIN, int WIN, int O>
__device__ __forceinline__ void kan_position(
    const float* __restrict__ in_b,   // [CIN, HIN, WIN] for this batch elem
    const float* __restrict__ wb,     // [O, F]
    const float* __restrict__ ws,     // [O, F, 8]
    int h0, int w0, float* acc)       // acc[O]
{
  constexpr int F = CIN * KS * KS;
  for (int c = 0; c < CIN; ++c) {
    for (int ky = 0; ky < KS; ++ky) {
      for (int kx = 0; kx < KS; ++kx) {
        const int f = (c * KS + ky) * KS + kx;      // patch feature index
        const float x = in_b[(c * HIN + h0 + ky) * WIN + w0 + kx];

        // SiLU
        const float sil = x / (1.0f + __expf(-x));

        // basis interval + local coordinate
        const float t  = (x + 2.2f) * 2.5f;
        const float ff = floorf(t);
        const int   i  = (int)ff;
        const float u  = t - ff;
        const float u2 = u * u, u3 = u2 * u;
        const float om = 1.0f - u;
        const float b0 = om * om * om * (1.0f / 6.0f);
        const float b1 = (3.0f * u3 - 6.0f * u2 + 4.0f) * (1.0f / 6.0f);
        const float b2 = (-3.0f * u3 + 3.0f * u2 + 3.0f * u + 1.0f) * (1.0f / 6.0f);
        const float b3 = u3 * (1.0f / 6.0f);

        // scatter the 4 taps into a dense 8-vector via unrolled selects
        // (x outside [-2.2,2.2) => i outside 0..10 => all selects miss => 0,
        //  matching the reference's out-of-grid zero basis)
        const int base = i - 3;                      // tap 0 lands at j=base
        float s[8];
        #pragma unroll
        for (int j = 0; j < 8; ++j) {
          float v = 0.0f;
          v = (base == j    ) ? b0 : v;
          v = (base == j - 1) ? b1 : v;
          v = (base == j - 2) ? b2 : v;
          v = (base == j - 3) ? b3 : v;
          s[j] = v;
        }

        // weight addresses are wave-uniform (depend only on loop counters)
        // -> scalar loads through the constant cache
        #pragma unroll
        for (int o = 0; o < O; ++o) {
          float a = sil * wb[o * F + f];
          const float* wr = ws + (size_t)(o * F + f) * 8;
          #pragma unroll
          for (int j = 0; j < 8; ++j) a += s[j] * wr[j];
          acc[o] += a;
        }
      }
    }
  }
}

// conv1 (1->5, k=3, 28x28 -> 26x26) fused with 2x2 maxpool -> [B,5,13,13]
__global__ __launch_bounds__(256)
void kan1_pool_kernel(const float* __restrict__ x,
                      const float* __restrict__ wb1,
                      const float* __restrict__ ws1,
                      float* __restrict__ h1p, int total)
{
  const int tid = blockIdx.x * blockDim.x + threadIdx.x;
  if (tid >= total) return;
  const int b  = tid / 169;
  const int r  = tid % 169;
  const int ph = r / 13, pw = r % 13;
  const float* xb = x + (size_t)b * 784;

  float best[5] = {-3.402823466e38f, -3.402823466e38f, -3.402823466e38f,
                   -3.402823466e38f, -3.402823466e38f};
  for (int dy = 0; dy < 2; ++dy) {
    for (int dx = 0; dx < 2; ++dx) {
      float acc[5] = {0.f, 0.f, 0.f, 0.f, 0.f};
      kan_position<1, 3, 28, 28, 5>(xb, wb1, ws1, 2 * ph + dy, 2 * pw + dx, acc);
      #pragma unroll
      for (int o = 0; o < 5; ++o) best[o] = fmaxf(best[o], acc[o]);
    }
  }
  #pragma unroll
  for (int o = 0; o < 5; ++o)
    h1p[(((size_t)b * 5 + o) * 13 + ph) * 13 + pw] = best[o];
}

// conv2 (5->5, k=5, 13x13 -> 9x9)
__global__ __launch_bounds__(256)
void kan2_kernel(const float* __restrict__ h1p,
                 const float* __restrict__ wb2,
                 const float* __restrict__ ws2,
                 float* __restrict__ h2, int total)
{
  const int tid = blockIdx.x * blockDim.x + threadIdx.x;
  if (tid >= total) return;
  const int b  = tid / 81;
  const int r  = tid % 81;
  const int ho = r / 9, wo = r % 9;
  const float* in_b = h1p + (size_t)b * 5 * 13 * 13;

  float acc[5] = {0.f, 0.f, 0.f, 0.f, 0.f};
  kan_position<5, 5, 13, 13, 5>(in_b, wb2, ws2, ho, wo, acc);
  #pragma unroll
  for (int o = 0; o < 5; ++o)
    h2[(((size_t)b * 5 + o) * 9 + ho) * 9 + wo] = acc[o];
}

// conv3 (5->2, k=3, 9x9 -> 7x7): writes the FC A-matrix directly as
// zero-padded f16 [B, 128] (valid cols 0..97 = c*49 + h*7 + w).
__global__ __launch_bounds__(256)
void kan3_kernel(const float* __restrict__ h2,
                 const float* __restrict__ wb3,
                 const float* __restrict__ ws3,
                 _Float16* __restrict__ Ah, int total)
{
  const int tid = blockIdx.x * blockDim.x + threadIdx.x;
  if (tid >= total) return;
  const int b  = tid / 49;
  const int r  = tid % 49;
  const int ho = r / 7, wo = r % 7;
  const float* in_b = h2 + (size_t)b * 5 * 9 * 9;

  float acc[2] = {0.f, 0.f};
  kan_position<5, 3, 9, 9, 2>(in_b, wb3, ws3, ho, wo, acc);
  #pragma unroll
  for (int o = 0; o < 2; ++o)
    Ah[(size_t)b * 128 + o * 49 + ho * 7 + wo] = (_Float16)acc[o];
}

// zero the K-padding columns (98..127 = dwords 49..63) of Ah [4096,128] f16
__global__ __launch_bounds__(256)
void pad_a_kernel(unsigned int* __restrict__ Ah_u32, int total)
{
  const int tid = blockIdx.x * blockDim.x + threadIdx.x;
  if (tid >= total) return;                 // total = 4096 * 15 dwords
  const int row = tid / 15, d = tid % 15;
  Ah_u32[(size_t)row * 64 + 49 + d] = 0u;
}

// pack fc_w [200,98] f32 -> Bh [208,128] f16, zero padded
__global__ __launch_bounds__(256)
void pack_b_kernel(const float* __restrict__ fc_w,
                   _Float16* __restrict__ Bh, int total)
{
  const int tid = blockIdx.x * blockDim.x + threadIdx.x;
  if (tid >= total) return;                 // total = 208 * 128
  const int n = tid / 128, k = tid % 128;
  const float v = (n < 200 && k < 98) ? fc_w[(size_t)n * 98 + k] : 0.0f;
  Bh[tid] = (_Float16)v;
}

// FC: out[4096,200] = Ah[4096,128] @ Bh[208,128]^T + fc_b
// One wave per 16x16 output tile; 4 WMMA steps over K=128; all fragment
// loads are unguarded, aligned wide loads (global_load_b128).
__global__ __launch_bounds__(32)
void fc_wmma_kernel(const _Float16* __restrict__ Ah,
                    const _Float16* __restrict__ Bh,
                    const float* __restrict__ fc_b,
                    float* __restrict__ out)
{
  const int lane = threadIdx.x;         // wave32 lane
  const int m0   = blockIdx.x * 16;     // 0..4080
  const int n0   = blockIdx.y * 16;     // 0..192 (13th tile padded in Bh)
  const int l15  = lane & 15;
  const bool hi  = (lane & 16) != 0;

  const _Float16* arow = Ah + (size_t)(m0 + l15) * 128;
  const _Float16* brow = Bh + (size_t)(n0 + l15) * 128;

  v8f c = {};
  #pragma unroll
  for (int kk = 0; kk < 128; kk += 32) {
    // A fragment (16x32 f16): lanes 0-15 hold K kk+0..15;
    // lanes 16-31 hold K kk+16..31 (same k order as B).
    const int ka = kk + (hi ? 16 : 0);
    union { v16h v; v8h h[2]; } a;
    a.h[0] = *(const v8h*)(arow + ka);        // 16B aligned
    a.h[1] = *(const v8h*)(arow + ka + 8);    // 16B aligned
    // B fragment (32x16 f16): lane holds column (lane&15); lanes 0-15 carry
    // K kk+0..15, lanes 16-31 carry K kk+16..31 -> one contiguous 32B chunk.
    const int kb = kk + (hi ? 16 : 0);
    const v16h b = *(const v16h*)(brow + kb); // 32B aligned
    // (neg_a, A, neg_b, B, c_mod, C, reuse_a, reuse_b)
    c = __builtin_amdgcn_wmma_f32_16x16x32_f16(false, a.v, false, b,
                                               (short)0, c, false, false);
  }

  // D layout: VGPR r -> M = r (+8 for lanes 16-31), N = lane&15.
  const int col = n0 + l15;
  if (col < 200) {
    const float bias = fc_b[col];
    #pragma unroll
    for (int r = 0; r < 8; ++r)
      out[(size_t)(m0 + r + (hi ? 8 : 0)) * 200 + col] = c[r] + bias;
  }
}

extern "C" void kernel_launch(void* const* d_in, const int* in_sizes, int n_in,
                              void* d_out, int out_size, void* d_ws, size_t ws_size,
                              hipStream_t stream) {
  (void)in_sizes; (void)n_in; (void)out_size; (void)ws_size;

  const float* x    = (const float*)d_in[0];  // [4096,1,28,28]
  const float* wb1  = (const float*)d_in[1];  // [5,9]
  const float* ws1  = (const float*)d_in[2];  // [5,9,8]
  const float* wb2  = (const float*)d_in[3];  // [5,125]
  const float* ws2  = (const float*)d_in[4];  // [5,125,8]
  const float* wb3  = (const float*)d_in[5];  // [2,45]
  const float* ws3  = (const float*)d_in[6];  // [2,45,8]
  const float* fcw  = (const float*)d_in[7];  // [200,98]
  const float* fcb  = (const float*)d_in[8];  // [200]
  float*       out  = (float*)d_out;          // [4096,200]

  // scratch carve-out:
  //   h1p f32 [4096,5,13,13]  = 3,461,120 floats (13.84 MB)
  //   h2  f32 [4096,5,9,9]    = 1,658,880 floats ( 6.64 MB)
  //   Ah  f16 [4096,128]      =   524,288 halves ( 1.00 MB)
  //   Bh  f16 [208,128]       =    26,624 halves ( 53 KB)     (~21.6 MB total)
  float*    h1p = (float*)d_ws;
  float*    h2  = h1p + (size_t)4096 * 5 * 13 * 13;
  _Float16* Ah  = (_Float16*)(h2 + (size_t)4096 * 5 * 9 * 9);
  _Float16* Bh  = Ah + (size_t)4096 * 128;

  const int B = 4096;

  {
    const int total = B * 13 * 13;
    kan1_pool_kernel<<<(total + 255) / 256, 256, 0, stream>>>(x, wb1, ws1, h1p, total);
  }
  {
    const int total = B * 9 * 9;
    kan2_kernel<<<(total + 255) / 256, 256, 0, stream>>>(h1p, wb2, ws2, h2, total);
  }
  {
    const int total = B * 15;  // zero pad dwords of Ah
    pad_a_kernel<<<(total + 255) / 256, 256, 0, stream>>>((unsigned int*)Ah, total);
  }
  {
    const int total = 208 * 128;  // pack/pad fc_w -> Bh
    pack_b_kernel<<<(total + 255) / 256, 256, 0, stream>>>(fcw, Bh, total);
  }
  {
    const int total = B * 7 * 7;
    kan3_kernel<<<(total + 255) / 256, 256, 0, stream>>>(h2, wb3, ws3, Ah, total);
  }
  {
    dim3 grid(B / 16, (200 + 15) / 16);  // 256 x 13 tiles, one wave each
    fc_wmma_kernel<<<grid, 32, 0, stream>>>(Ah, Bh, fcb, out);
  }
}
